// GATBase_82489141887125
// MI455X (gfx1250) — compile-verified, matrix operand order checked by reading, not run on hardware
//
#include <hip/hip_runtime.h>

#define NN 100000
#define EE 1600000
#define DD 64
#define LL 4
#define HH 4
#define CC 16
#define DIN 192

// padded LDS row strides (avoid bank conflicts; keep 16B alignment for b128 frags)
#define ASTRIDE 200   // bf16 elems per row, edge kernel A stage  (400 B/row)
#define HSTRIDE 196   // f32  elems per row, edge kernel H buffer (784 B/row)
#define NSTRIDE 72    // bf16 elems per row, node kernel A stage  (144 B/row)

typedef __bf16 bf16_t;
typedef __attribute__((ext_vector_type(16))) __bf16 v16bf;
typedef __attribute__((ext_vector_type(2)))  __bf16 v2bf;
typedef __attribute__((ext_vector_type(8)))  float  v8f;

static __device__ inline v8f wmma_bf16(v16bf a, v16bf b, v8f c) {
  // D = A(16x32) * B(32x16) + C, f32 accumulate
  return __builtin_amdgcn_wmma_f32_16x16x32_bf16(false, a, false, b, (short)0, c, false, false);
}

// Fragment slice for this lane: elements 0..7 at [k0..k0+7], 8..15 at [k0+16..k0+23]
// (two contiguous 16B runs -> b128 loads).
static __device__ inline v16bf load16(const bf16_t* base, int k0) {
  v16bf v;
#pragma unroll
  for (int i = 0; i < 8; ++i) { v[i] = base[k0 + i]; v[i + 8] = base[k0 + 16 + i]; }
  return v;
}

static __device__ inline v2bf pk2(float x, float y) {
  v2bf r; r[0] = (bf16_t)x; r[1] = (bf16_t)y; return r;   // -> v_cvt_pk_bf16_f32
}

// monotonic float<->int map for atomicMax on floats
static __device__ inline int   mapf(float f)  { int i = __float_as_int(f); return i >= 0 ? i : (i ^ 0x7fffffff); }
static __device__ inline float unmapf(int m)  { int i = m >= 0 ? m : (m ^ 0x7fffffff); return __int_as_float(i); }

// ---------------- per-layer weight prep: bf16 + transpose, coalesced writes -----------------
__global__ __launch_bounds__(256) void convert_weights_kernel(
    const float* __restrict__ cw, const float* __restrict__ ew,
    const float* __restrict__ w1, const float* __restrict__ w2,
    bf16_t* __restrict__ cwt, bf16_t* __restrict__ ewt,
    bf16_t* __restrict__ w1t, bf16_t* __restrict__ w2t)
{
  int i = blockIdx.x * 256 + threadIdx.x;
  if (i < DD * DD)  { int n = i >> 6, k = i & 63;   cwt[i] = (bf16_t)cw[k * DD + n];
                                                    ewt[i] = (bf16_t)ew[k * DD + n]; }
  if (i < DIN * DIN){ int n = i / DIN, k = i % DIN; w1t[i] = (bf16_t)w1[k * DIN + n]; }
  if (i < DIN * DD) { int n = i / DIN, k = i % DIN; w2t[i] = (bf16_t)w2[(size_t)k * DD + n]; }
}

// ---------------- node projection: xh = nf @ conv_w  (WMMA bf16) ----------------------------
__global__ __launch_bounds__(256) void node_proj_kernel(
    const float* __restrict__ nf, const bf16_t* __restrict__ cwt,
    float* __restrict__ xh, int ntiles)
{
  __shared__ __align__(16) bf16_t sA[8][16 * NSTRIDE];
  const int wave = threadIdx.x >> 5, lane = threadIdx.x & 31;
  const int tile = blockIdx.x * 8 + wave;
  if (tile >= ntiles) return;
  bf16_t* A = sA[wave];
  const int m = lane & 15, half = lane >> 4, base = tile * 16;

  // stage: each lane packs 2 consecutive columns -> one b32 LDS store per row
#pragma unroll 4
  for (int r = 0; r < 16; ++r) {
    const float2* p = (const float2*)(nf + (size_t)(base + r) * DD);
    float2 x = p[lane];
    ((v2bf*)(A + r * NSTRIDE))[lane] = pk2(x.x, x.y);
  }
  v16bf A0 = load16(A + m * NSTRIDE, 8 * half);
  v16bf A1 = load16(A + m * NSTRIDE, 32 + 8 * half);
#pragma unroll
  for (int nt = 0; nt < 4; ++nt) {
    const int col = nt * 16 + m;
    const bf16_t* wcol = cwt + (size_t)col * DD;
    v8f acc;
#pragma unroll
    for (int r = 0; r < 8; ++r) acc[r] = 0.f;
    acc = wmma_bf16(A0, load16(wcol, 8 * half), acc);
    acc = wmma_bf16(A1, load16(wcol, 32 + 8 * half), acc);
#pragma unroll
    for (int r = 0; r < 8; ++r)
      xh[(size_t)(base + half * 8 + r) * DD + col] = acc[r];
  }
}

// ---------------- alpha_src / alpha_dst from xh ---------------------------------------------
__global__ __launch_bounds__(256) void alpha_nodes_kernel(
    const float* __restrict__ xh, const float* __restrict__ asw, const float* __restrict__ adw,
    float* __restrict__ asrc, float* __restrict__ adst)
{
  int i = blockIdx.x * 256 + threadIdx.x;
  if (i >= NN * HH) return;
  int n = i >> 2, h = i & 3;
  const float4* xr = (const float4*)(xh + (size_t)n * DD + h * CC);
  float a = 0.f, b = 0.f;
#pragma unroll
  for (int c4 = 0; c4 < 4; ++c4) {
    float4 x = xr[c4];
    const float* ws = asw + h * CC + c4 * 4;
    const float* wd = adw + h * CC + c4 * 4;
    a += x.x * ws[0] + x.y * ws[1] + x.z * ws[2] + x.w * ws[3];
    b += x.x * wd[0] + x.y * wd[1] + x.z * wd[2] + x.w * wd[3];
  }
  asrc[i] = a; adst[i] = b;
}

// ---------------- fold edge_w with att_edge: v[d][h] ----------------------------------------
__global__ void vpre_kernel(const float* __restrict__ ew, const float* __restrict__ aew,
                            float* __restrict__ vpre)
{
  int i = threadIdx.x;               // 256 = 64*4
  int d = i >> 2, h = i & 3;
  float s = 0.f;
#pragma unroll
  for (int c = 0; c < CC; ++c) s += ew[d * DD + h * CC + c] * aew[h * CC + c];
  vpre[i] = s;
}

// ---------------- alpha_edge = ef @ v --------------------------------------------------------
__global__ __launch_bounds__(256) void alpha_edge_kernel(
    const float* __restrict__ ef, const float* __restrict__ vpre, float* __restrict__ ae)
{
  __shared__ float sv[DD * HH];
  for (int j = threadIdx.x; j < DD * HH; j += 256) sv[j] = vpre[j];
  __syncthreads();
  int e = blockIdx.x * 256 + threadIdx.x;
  if (e >= EE) return;
  const float4* pe = (const float4*)(ef + (size_t)e * DD);
  float a0 = 0.f, a1 = 0.f, a2 = 0.f, a3 = 0.f;
#pragma unroll 4
  for (int d4 = 0; d4 < 16; ++d4) {
    float4 x = pe[d4];
    const float* v0 = sv + (d4 * 4 + 0) * 4;
    a0 += x.x * v0[0] + x.y * v0[4] + x.z * v0[8]  + x.w * v0[12];
    a1 += x.x * v0[1] + x.y * v0[5] + x.z * v0[9]  + x.w * v0[13];
    a2 += x.x * v0[2] + x.y * v0[6] + x.z * v0[10] + x.w * v0[14];
    a3 += x.x * v0[3] + x.y * v0[7] + x.z * v0[11] + x.w * v0[15];
  }
  ae[(size_t)e * 4 + 0] = a0; ae[(size_t)e * 4 + 1] = a1;
  ae[(size_t)e * 4 + 2] = a2; ae[(size_t)e * 4 + 3] = a3;
}

// ---------------- attention scratch init ----------------------------------------------------
__global__ void att_init_kernel(int* __restrict__ amaxi, float* __restrict__ asum,
                                float* __restrict__ agg)
{
  int i = blockIdx.x * 256 + threadIdx.x;
  if (i < NN * DD) agg[i] = 0.f;
  if (i < NN * HH) { amaxi[i] = (int)0x80000000u; asum[i] = 0.f; }
}

// ---------------- pass A: leaky-relu pre-activation + segment max ----------------------------
__global__ __launch_bounds__(256) void att_pass_a(
    const int* __restrict__ srcI, const int* __restrict__ dstI,
    const float* __restrict__ asrc, const float* __restrict__ adst,
    float* __restrict__ ae, int* __restrict__ amaxi)
{
  int e = blockIdx.x * 256 + threadIdx.x;
  if (e >= EE) return;
  int s = srcI[e], d = dstI[e];
#pragma unroll
  for (int h = 0; h < HH; ++h) {
    float a = asrc[s * 4 + h] + adst[d * 4 + h] + ae[(size_t)e * 4 + h];
    a = a > 0.f ? a : 0.2f * a;
    ae[(size_t)e * 4 + h] = a;
    atomicMax(&amaxi[d * 4 + h], mapf(a));
  }
}
__global__ void att_pass_a_self(const float* __restrict__ asrc, const float* __restrict__ adst,
                                float* __restrict__ aself, int* __restrict__ amaxi)
{
  int i = blockIdx.x * 256 + threadIdx.x;
  if (i >= NN * HH) return;
  float a = asrc[i] + adst[i];
  a = a > 0.f ? a : 0.2f * a;
  aself[i] = a;
  atomicMax(&amaxi[i], mapf(a));
}

// ---------------- pass B: exp + segment sum --------------------------------------------------
__global__ __launch_bounds__(256) void att_pass_b(
    const int* __restrict__ dstI, float* __restrict__ ae,
    const int* __restrict__ amaxi, float* __restrict__ asum)
{
  int e = blockIdx.x * 256 + threadIdx.x;
  if (e >= EE) return;
  int d = dstI[e];
#pragma unroll
  for (int h = 0; h < HH; ++h) {
    int idx = d * 4 + h;
    float a = expf(ae[(size_t)e * 4 + h] - unmapf(amaxi[idx]));
    ae[(size_t)e * 4 + h] = a;
    atomicAdd(&asum[idx], a);
  }
}
__global__ void att_pass_b_self(float* __restrict__ aself, const int* __restrict__ amaxi,
                                float* __restrict__ asum)
{
  int i = blockIdx.x * 256 + threadIdx.x;
  if (i >= NN * HH) return;
  float a = expf(aself[i] - unmapf(amaxi[i]));
  aself[i] = a;
  atomicAdd(&asum[i], a);
}

// ---------------- pass C: normalized message scatter-add -------------------------------------
__global__ __launch_bounds__(256) void att_pass_c(
    const int* __restrict__ srcI, const int* __restrict__ dstI,
    const float* __restrict__ ae, const float* __restrict__ asum,
    const float* __restrict__ xh, float* __restrict__ agg)
{
  int i = blockIdx.x * 256 + threadIdx.x;
  if (i >= EE * HH) return;
  int e = i >> 2, h = i & 3;
  int s = srcI[e], d = dstI[e];
  float w = ae[i] / (asum[d * 4 + h] + 1e-16f);
  const float4* xr = (const float4*)(xh + (size_t)s * DD + h * CC);
  float* gr = agg + (size_t)d * DD + h * CC;
#pragma unroll
  for (int c4 = 0; c4 < 4; ++c4) {
    float4 x = xr[c4];
    atomicAdd(&gr[c4 * 4 + 0], x.x * w);
    atomicAdd(&gr[c4 * 4 + 1], x.y * w);
    atomicAdd(&gr[c4 * 4 + 2], x.z * w);
    atomicAdd(&gr[c4 * 4 + 3], x.w * w);
  }
}
__global__ void att_pass_c_self(const float* __restrict__ aself, const float* __restrict__ asum,
                                const float* __restrict__ xh, float* __restrict__ agg)
{
  int i = blockIdx.x * 256 + threadIdx.x;
  if (i >= NN * HH) return;
  int n = i >> 2, h = i & 3;
  float w = aself[i] / (asum[i] + 1e-16f);
  const float4* xr = (const float4*)(xh + (size_t)n * DD + h * CC);
  float* gr = agg + (size_t)n * DD + h * CC;
#pragma unroll
  for (int c4 = 0; c4 < 4; ++c4) {
    float4 x = xr[c4];
    atomicAdd(&gr[c4 * 4 + 0], x.x * w);
    atomicAdd(&gr[c4 * 4 + 1], x.y * w);
    atomicAdd(&gr[c4 * 4 + 2], x.z * w);
    atomicAdd(&gr[c4 * 4 + 3], x.w * w);
  }
}

// ---------------- node LayerNorm + ReLU + residual (wave per node, wave32) -------------------
__global__ __launch_bounds__(256) void node_ln_kernel(
    const float* __restrict__ agg, const float* __restrict__ cb,
    const float* __restrict__ g, const float* __restrict__ b,
    float* __restrict__ nf, int n_nodes)
{
  const int wave = threadIdx.x >> 5, lane = threadIdx.x & 31;
  const int n = blockIdx.x * 8 + wave;
  if (n >= n_nodes) return;
  const float2* row = (const float2*)(agg + (size_t)n * DD);
  float2 x = row[lane];
  int c0 = lane * 2, c1 = lane * 2 + 1;
  float x0 = x.x + cb[c0];
  float x1 = x.y + cb[c1];
  float s = x0 + x1, q = x0 * x0 + x1 * x1;
#pragma unroll
  for (int off = 1; off < 32; off <<= 1) { s += __shfl_xor(s, off); q += __shfl_xor(q, off); }
  float mean = s * (1.f / 64.f);
  float var  = q * (1.f / 64.f) - mean * mean;
  float inv  = rsqrtf(var + 1e-5f);
  float* out = nf + (size_t)n * DD;
  float y0 = fmaxf((x0 - mean) * inv * g[c0] + b[c0], 0.f);
  float y1 = fmaxf((x1 - mean) * inv * g[c1] + b[c1], 0.f);
  out[c0] = y0 + out[c0];
  out[c1] = y1 + out[c1];
}

// ---------------- edge-update MLP: the big WMMA kernel ---------------------------------------
// per wave: 16 edges. A=[nf[src]|nf[dst]|ef] (16x192 bf16 in LDS), GEMM1 -> f32 LDS,
// row LayerNorm+ReLU -> bf16 A2 (reuse stage), GEMM2 -> += b2 + ef (in place).
__global__ __launch_bounds__(64) void edge_update_kernel(
    const float* __restrict__ nf, float* __restrict__ ef,
    const int* __restrict__ srcI, const int* __restrict__ dstI,
    const bf16_t* __restrict__ w1t, const bf16_t* __restrict__ w2t,
    const float* __restrict__ b1, const float* __restrict__ lng,
    const float* __restrict__ lnb, const float* __restrict__ b2,
    int ntiles)
{
  __shared__ __align__(16) bf16_t sA[2][16 * ASTRIDE];
  __shared__ __align__(16) float  sH[2][16 * HSTRIDE];
  const int wave = threadIdx.x >> 5, lane = threadIdx.x & 31;
  const int tile = blockIdx.x * 2 + wave;
  if (tile >= ntiles) return;
  bf16_t* A  = sA[wave];
  float*  Hb = sH[wave];
  const int m = lane & 15, half = lane >> 4;
  const int base = tile * 16;

  // gather + pack f32->bf16x2 stage: one b32 LDS store per segment per row
#pragma unroll 4
  for (int r = 0; r < 16; ++r) {
    const int e = base + r;
    const float2* ps = (const float2*)(nf + (size_t)srcI[e] * DD);
    const float2* pd = (const float2*)(nf + (size_t)dstI[e] * DD);
    const float2* pe = (const float2*)(ef + (size_t)e * DD);
    float2 a = ps[lane], b = pd[lane], c = pe[lane];
    v2bf* row = (v2bf*)(A + r * ASTRIDE);
    row[lane]      = pk2(a.x, a.y);
    row[32 + lane] = pk2(b.x, b.y);
    row[64 + lane] = pk2(c.x, c.y);
  }

  // A fragments (K = 6 x 32)
  v16bf Areg[6];
#pragma unroll
  for (int ks = 0; ks < 6; ++ks) Areg[ks] = load16(A + m * ASTRIDE, ks * 32 + 8 * half);

  // GEMM1: H = A(16x192) @ W1(192x192) + b1
#pragma unroll 1
  for (int nt = 0; nt < 12; ++nt) {
    const int col = nt * 16 + m;
    const bf16_t* wcol = w1t + (size_t)col * DIN;
    float bias = b1[col];
    v8f acc;
#pragma unroll
    for (int r = 0; r < 8; ++r) acc[r] = bias;
#pragma unroll
    for (int ks = 0; ks < 6; ++ks)
      acc = wmma_bf16(Areg[ks], load16(wcol, ks * 32 + 8 * half), acc);
#pragma unroll
    for (int r = 0; r < 8; ++r) Hb[(half * 8 + r) * HSTRIDE + col] = acc[r];
  }

  // per-row LayerNorm(192) + ReLU -> bf16 A2 (reuse A stage)
  {
    float s = 0.f, q = 0.f;
    const int c0 = half * 96;
#pragma unroll 8
    for (int j = 0; j < 96; ++j) { float x = Hb[m * HSTRIDE + c0 + j]; s += x; q += x * x; }
    s += __shfl_xor(s, 16);
    q += __shfl_xor(q, 16);
    float mean = s * (1.f / 192.f);
    float var  = q * (1.f / 192.f) - mean * mean;
    float inv  = rsqrtf(var + 1e-5f);
#pragma unroll 4
    for (int j = 0; j < 96; j += 2) {
      int col = c0 + j;
      float xa = Hb[m * HSTRIDE + col];
      float xb = Hb[m * HSTRIDE + col + 1];
      float ya = fmaxf((xa - mean) * inv * lng[col] + lnb[col], 0.f);
      float yb = fmaxf((xb - mean) * inv * lng[col + 1] + lnb[col + 1], 0.f);
      *(v2bf*)(A + m * ASTRIDE + col) = pk2(ya, yb);   // col even, 4B aligned
    }
  }

  // A2 fragments
  v16bf A2[6];
#pragma unroll
  for (int ks = 0; ks < 6; ++ks) A2[ks] = load16(A + m * ASTRIDE, ks * 32 + 8 * half);

  // GEMM2: ef += H2(16x192) @ W2(192x64) + b2
#pragma unroll 1
  for (int nt = 0; nt < 4; ++nt) {
    const int col = nt * 16 + m;
    const bf16_t* wcol = w2t + (size_t)col * DIN;
    v8f acc;
#pragma unroll
    for (int r = 0; r < 8; ++r) acc[r] = 0.f;
#pragma unroll
    for (int ks = 0; ks < 6; ++ks)
      acc = wmma_bf16(A2[ks], load16(wcol, ks * 32 + 8 * half), acc);
    float bias = b2[col];
#pragma unroll
    for (int r = 0; r < 8; ++r) {
      float* p = ef + (size_t)(base + half * 8 + r) * DD + col;
      *p = *p + acc[r] + bias;
    }
  }
}

// ================================ host orchestration =========================================
extern "C" void kernel_launch(void* const* d_in, const int* in_sizes, int n_in,
                              void* d_out, int out_size, void* d_ws, size_t ws_size,
                              hipStream_t stream)
{
  (void)in_sizes; (void)n_in; (void)out_size; (void)ws_size;
  const float* node_feats = (const float*)d_in[0];
  const float* edge_feats = (const float*)d_in[1];
  const int*   edge_index = (const int*)  d_in[2];
  const float* conv_w   = (const float*)d_in[3];
  const float* att_src  = (const float*)d_in[4];
  const float* att_dst  = (const float*)d_in[5];
  const float* edge_w   = (const float*)d_in[6];
  const float* att_edge = (const float*)d_in[7];
  const float* conv_b   = (const float*)d_in[8];
  const float* ln_g     = (const float*)d_in[9];
  const float* ln_b     = (const float*)d_in[10];
  const float* up1_w    = (const float*)d_in[11];
  const float* up1_b    = (const float*)d_in[12];
  const float* up_ln_g  = (const float*)d_in[13];
  const float* up_ln_b  = (const float*)d_in[14];
  const float* up2_w    = (const float*)d_in[15];
  const float* up2_b    = (const float*)d_in[16];

  float* nf = (float*)d_out;                 // output nf lives in d_out[0 : N*64)
  float* ef = nf + (size_t)NN * DD;          // output ef lives right after
  const int* srcI = edge_index;
  const int* dstI = edge_index + EE;

  char* wsp = (char*)d_ws;
  size_t off = 0;
  auto carve = [&](size_t bytes) -> char* {
    char* p = wsp + off; off += (bytes + 255) & ~(size_t)255; return p;
  };
  float*  xh    = (float*) carve((size_t)NN * DD * 4);
  float*  asrc  = (float*) carve((size_t)NN * HH * 4);
  float*  adst  = (float*) carve((size_t)NN * HH * 4);
  float*  ae    = (float*) carve((size_t)EE * HH * 4);
  float*  aself = (float*) carve((size_t)NN * HH * 4);
  int*    amaxi = (int*)   carve((size_t)NN * HH * 4);
  float*  asum  = (float*) carve((size_t)NN * HH * 4);
  float*  agg   = (float*) carve((size_t)NN * DD * 4);
  float*  vpre  = (float*) carve((size_t)DD * HH * 4);
  bf16_t* cwt   = (bf16_t*)carve((size_t)DD * DD * 2);
  bf16_t* ewt   = (bf16_t*)carve((size_t)DD * DD * 2);
  bf16_t* w1t   = (bf16_t*)carve((size_t)DIN * DIN * 2);
  bf16_t* w2t   = (bf16_t*)carve((size_t)DIN * DD * 2);

  hipMemcpyAsync(nf, node_feats, (size_t)NN * DD * 4, hipMemcpyDeviceToDevice, stream);
  hipMemcpyAsync(ef, edge_feats, (size_t)EE * DD * 4, hipMemcpyDeviceToDevice, stream);

  const int ntilesN = NN / 16;   // 6250
  const int ntilesE = EE / 16;   // 100000

  for (int l = 0; l < LL; ++l) {
    const float* cw_l = conv_w + (size_t)l * DD * DD;
    const float* ew_l = edge_w + (size_t)l * DD * DD;
    const float* w1_l = up1_w  + (size_t)l * DIN * DIN;
    const float* w2_l = up2_w  + (size_t)l * DIN * DD;

    convert_weights_kernel<<<(DIN * DIN + 255) / 256, 256, 0, stream>>>(
        cw_l, ew_l, w1_l, w2_l, cwt, ewt, w1t, w2t);

    node_proj_kernel<<<(ntilesN + 7) / 8, 256, 0, stream>>>(nf, cwt, xh, ntilesN);

    alpha_nodes_kernel<<<(NN * HH + 255) / 256, 256, 0, stream>>>(
        xh, att_src + (size_t)l * HH * CC, att_dst + (size_t)l * HH * CC, asrc, adst);

    vpre_kernel<<<1, 256, 0, stream>>>(ew_l, att_edge + (size_t)l * HH * CC, vpre);

    alpha_edge_kernel<<<(EE + 255) / 256, 256, 0, stream>>>(ef, vpre, ae);

    att_init_kernel<<<(NN * DD + 255) / 256, 256, 0, stream>>>(amaxi, asum, agg);

    att_pass_a<<<(EE + 255) / 256, 256, 0, stream>>>(srcI, dstI, asrc, adst, ae, amaxi);
    att_pass_a_self<<<(NN * HH + 255) / 256, 256, 0, stream>>>(asrc, adst, aself, amaxi);

    att_pass_b<<<(EE + 255) / 256, 256, 0, stream>>>(dstI, ae, amaxi, asum);
    att_pass_b_self<<<(NN * HH + 255) / 256, 256, 0, stream>>>(aself, amaxi, asum);

    att_pass_c<<<(EE * HH + 255) / 256, 256, 0, stream>>>(srcI, dstI, ae, asum, xh, agg);
    att_pass_c_self<<<(NN * HH + 255) / 256, 256, 0, stream>>>(aself, asum, xh, agg);

    node_ln_kernel<<<(NN + 7) / 8, 256, 0, stream>>>(
        agg, conv_b + (size_t)l * DD, ln_g + (size_t)l * DD, ln_b + (size_t)l * DD, nf, NN);

    edge_update_kernel<<<(ntilesE + 1) / 2, 64, 0, stream>>>(
        nf, ef, srcI, dstI, w1t, w2t,
        up1_b + (size_t)l * DIN, up_ln_g + (size_t)l * DIN,
        up_ln_b + (size_t)l * DIN, up2_b + (size_t)l * DD, ntilesE);
  }
}
